// NeuralParser_50757923504402
// MI455X (gfx1250) — matrix operand order repaired and checked by reading, not compiled, not run on hardware
//
#include <hip/hip_runtime.h>
#include <hip/hip_bf16.h>

// ---- problem dims ----
#define BB 512
#define PP 256
#define TT 128
#define EE 300
#define HH 256
#define NCC 64
#define NOO 256
#define KL 576   // padded E+H (556 -> 576)
#define KG 864   // padded H+2E (856 -> 864)
#define KH 256   // gru hh K
#define NL 1024  // 4H
#define NG 768   // 3H

typedef __attribute__((ext_vector_type(16))) _Float16 v16h;
typedef __attribute__((ext_vector_type(8)))  _Float16 v8h;
typedef __attribute__((ext_vector_type(8)))  float    v8f;

__device__ __forceinline__ float sigmoidf_(float x) { return 1.0f / (1.0f + __expf(-x)); }

// A fragment: 16x32 f16, row-major A with leading dim ldA.
// Lane L<16: row m0+L, K = k0+{0..15}
// Lane L>=16: row m0+L-16, K = k0+{16..31}
__device__ __forceinline__ v16h load_a_frag(const _Float16* __restrict__ A, int ldA,
                                            int m0, int k0, int lane) {
  int half = lane >> 4;
  int l    = lane & 15;
  const _Float16* p = A + (size_t)(m0 + l) * ldA + k0 + half * 16;
  v8h lo = *(const v8h*)(p);
  v8h hi = *(const v8h*)(p + 8);
  v16h a;
#pragma unroll
  for (int i = 0; i < 8; ++i) { a[i] = lo[i]; a[i + 8] = hi[i]; }
  return a;
}

// B fragment from pre-packed layout Bp[kk][n][16], kk = K/16 chunk.
// Lane L<16: col n0+L, K rows k0..k0+15 ; Lane L>=16: col n0+L-16, K rows k0+16..k0+31
__device__ __forceinline__ v16h load_b_frag(const _Float16* __restrict__ Bp, int N,
                                            int n0, int k0, int lane) {
  int half = lane >> 4;
  int l    = lane & 15;
  const _Float16* p = Bp + ((size_t)((k0 >> 4) + half) * N + (n0 + l)) * 16;
  return *(const v16h*)p;
}

// Multi-gate, dual-M-tile fragment GEMM: one wave computes a 32x16 (m,h) patch
// for G gate matrices. Each B fragment load feeds two independent WMMAs (one
// per M-tile), halving B traffic per WMMA and giving the scheduler a no-wait
// WMMA to overlap with the next load's latency.
template <int K, int G>
__device__ __forceinline__ void gemm_gates2(const _Float16* __restrict__ A, int ldA,
                                            const _Float16* __restrict__ Bp, int N,
                                            int m0, int h0, int lane, v8f (*acc)[2]) {
#pragma unroll
  for (int k0 = 0; k0 < K; k0 += 32) {
    v16h a0 = load_a_frag(A, ldA, m0, k0, lane);
    v16h a1 = load_a_frag(A, ldA, m0 + 16, k0, lane);
#pragma unroll
    for (int g = 0; g < G; ++g) {
      v16h b = load_b_frag(Bp, N, g * HH + h0, k0, lane);
      acc[g][0] = __builtin_amdgcn_wmma_f32_16x16x32_f16(false, a0, false, b,
                                                         (short)0, acc[g][0], false, false);
      acc[g][1] = __builtin_amdgcn_wmma_f32_16x16x32_f16(false, a1, false, b,
                                                         (short)0, acc[g][1], false, false);
    }
  }
}

// ---- weight pre-pack: W1 [N,K1] (+optional W2 [N,K2] concatenated on K) -> Bp[Kpad/16][N][16] f16
__global__ void pack_b2_kernel(const float* __restrict__ W1, int K1,
                               const float* __restrict__ W2, int K2,
                               _Float16* __restrict__ Bp, int N, int Kpad) {
  int e = blockIdx.x * blockDim.x + threadIdx.x;
  int total = Kpad * N;
  if (e >= total) return;
  int j  = e & 15;
  int r  = e >> 4;
  int n  = r % N;
  int kk = r / N;
  int k  = kk * 16 + j;
  float v = 0.0f;
  if (k < K1) v = W1[(size_t)n * K1 + k];
  else if (W2 != nullptr && k < K1 + K2) v = W2[(size_t)n * K2 + (k - K1)];
  Bp[e] = (_Float16)v;
}

// ---- encoder: pack A row [emb(tok) | h | 0pad] as f16
__global__ void pack_x_kernel(const float* __restrict__ emb, const int* __restrict__ program,
                              const float* __restrict__ h, _Float16* __restrict__ Aq, int t) {
  int b = blockIdx.x;
  int tok = program[b * PP + t];
  const float* er = emb + (size_t)tok * EE;
  for (int j = threadIdx.x; j < KL; j += blockDim.x) {
    float v = 0.0f;
    if (j < EE)            v = er[j];
    else if (j < EE + HH)  v = h[b * HH + (j - EE)];
    Aq[(size_t)b * KL + j] = (_Float16)v;
  }
}

// ---- encoder: fused gates-GEMM (WMMA) + LSTM update.
// One wave = 32 batch rows x 16 h-cols, all 4 gates.
__global__ void lstm_step_kernel(const _Float16* __restrict__ Aq, const _Float16* __restrict__ Bp,
                                 const float* __restrict__ bih, const float* __restrict__ bhh,
                                 const int* __restrict__ plen, int t,
                                 float* __restrict__ h, float* __restrict__ c) {
  int lane = threadIdx.x & 31;
  int gw   = blockIdx.x * (blockDim.x >> 5) + (threadIdx.x >> 5);
  int m0   = (gw >> 4) * 32;
  int h0   = (gw & 15) * 16;
  v8f acc[4][2] = {};
  gemm_gates2<KL, 4>(Aq, KL, Bp, NL, m0, h0, lane, acc);
  int colh = h0 + (lane & 15);
  float bi = bih[colh]          + bhh[colh];
  float bf = bih[HH + colh]     + bhh[HH + colh];
  float bg = bih[2 * HH + colh] + bhh[2 * HH + colh];
  float bo = bih[3 * HH + colh] + bhh[3 * HH + colh];
#pragma unroll
  for (int mt = 0; mt < 2; ++mt) {
    int rbase = m0 + mt * 16 + (lane >> 4) * 8;
#pragma unroll
    for (int v = 0; v < 8; ++v) {
      int row = rbase + v;
      if (t < plen[row]) {
        float iv = sigmoidf_(acc[0][mt][v] + bi);
        float fv = sigmoidf_(acc[1][mt][v] + bf);
        float gv = tanhf(acc[2][mt][v] + bg);
        float ov = sigmoidf_(acc[3][mt][v] + bo);
        size_t idx = (size_t)row * HH + colh;
        float cnew = fv * c[idx] + iv * gv;
        c[idx] = cnew;
        h[idx] = ov * tanhf(cnew);
      }
    }
  }
}

// ---- decoder: BatchNorm1d over batch (biased var), write f32 + f16 normalized h
__global__ void bn_kernel(const float* __restrict__ h, const float* __restrict__ gamma,
                          const float* __restrict__ beta,
                          float* __restrict__ hn, _Float16* __restrict__ hn16) {
  int lane = threadIdx.x & 31;
  int col  = blockIdx.x * 8 + (threadIdx.x >> 5);
  float s = 0.0f, s2 = 0.0f;
  for (int r = lane; r < BB; r += 32) {
    float v = h[(size_t)r * HH + col];
    s += v; s2 += v * v;
  }
#pragma unroll
  for (int off = 16; off > 0; off >>= 1) {
    s  += __shfl_xor(s, off, 32);
    s2 += __shfl_xor(s2, off, 32);
  }
  float mean = s * (1.0f / BB);
  float var  = s2 * (1.0f / BB) - mean * mean;
  float sc   = rsqrtf(var + 1e-5f) * gamma[col];
  float sh   = beta[col] - mean * sc;
  for (int r = lane; r < BB; r += 32) {
    size_t idx = (size_t)r * HH + col;
    float v = h[idx] * sc + sh;
    hn[idx]   = v;
    hn16[idx] = (_Float16)v;
  }
}

// ---- decoder: pack GRU input row [choice_emb | prog_emb | index_emb | 0pad] as f16
__global__ void pack_inp_kernel(const int* __restrict__ trace, const int* __restrict__ choices,
                                const float* __restrict__ ctab, const float* __restrict__ iemb,
                                const float* __restrict__ prog, _Float16* __restrict__ Iq, int t) {
  int b = blockIdx.x;
  int prev = trace[b * TT + t];
  int val  = choices[b * NCC + prev];
  const float* ch = ctab + ((size_t)prev * NOO + val) * EE;
  const float* ie = iemb + (size_t)prev * EE;
  for (int j = threadIdx.x; j < KG; j += blockDim.x) {
    float v = 0.0f;
    if (j < EE)                 v = ch[j];
    else if (j < EE + HH)       v = prog[b * HH + (j - EE)];
    else if (j < 2 * EE + HH)   v = ie[j - (EE + HH)];
    Iq[(size_t)b * KG + j] = (_Float16)v;
  }
}

// ---- decoder: fused dual-GEMM (inp@Wih, hn@Whh via WMMA) + GRU update.
// One wave = 32 batch rows x 16 h-cols, 3 gates from each GEMM.
__global__ void gru_step_kernel(const _Float16* __restrict__ Iq, const _Float16* __restrict__ BpIH,
                                const _Float16* __restrict__ Hn16, const _Float16* __restrict__ BpHH,
                                const float* __restrict__ hn, const float* __restrict__ bih,
                                const float* __restrict__ bhh, float* __restrict__ h) {
  int lane = threadIdx.x & 31;
  int gw   = blockIdx.x * (blockDim.x >> 5) + (threadIdx.x >> 5);
  int m0   = (gw >> 4) * 32;
  int h0   = (gw & 15) * 16;
  v8f ax[3][2] = {};
  v8f ah[3][2] = {};
  gemm_gates2<KG, 3>(Iq, KG, BpIH, NG, m0, h0, lane, ax);
  gemm_gates2<KH, 3>(Hn16, KH, BpHH, NG, m0, h0, lane, ah);
  int colh = h0 + (lane & 15);
  float brx = bih[colh], bzx = bih[HH + colh], bnx = bih[2 * HH + colh];
  float brh = bhh[colh], bzh = bhh[HH + colh], bnh = bhh[2 * HH + colh];
#pragma unroll
  for (int mt = 0; mt < 2; ++mt) {
    int rbase = m0 + mt * 16 + (lane >> 4) * 8;
#pragma unroll
    for (int v = 0; v < 8; ++v) {
      int row = rbase + v;
      float r = sigmoidf_(ax[0][mt][v] + brx + ah[0][mt][v] + brh);
      float z = sigmoidf_(ax[1][mt][v] + bzx + ah[1][mt][v] + bzh);
      float n = tanhf(ax[2][mt][v] + bnx + r * (ah[2][mt][v] + bnh));
      size_t idx = (size_t)row * HH + colh;
      h[idx] = (1.0f - z) * n + z * hn[idx];
    }
  }
}

// ---- decoder: routed per-sample linear head + not_ended mask
__global__ void head_kernel(const int* __restrict__ trace, const float* __restrict__ h,
                            const float* __restrict__ infW, const float* __restrict__ infb,
                            float* __restrict__ out, int t) {
  int b = blockIdx.x;
  int o = threadIdx.x;
  int cur = trace[b * TT + t + 1];
  __shared__ float hs[HH];
  hs[o] = h[(size_t)b * HH + o];
  __syncthreads();
  const float4* w = (const float4*)(infW + ((size_t)cur * NOO + o) * HH);
  float acc = infb[cur * NOO + o];
#pragma unroll 4
  for (int k = 0; k < HH / 4; ++k) {
    float4 wv = w[k];
    acc += wv.x * hs[4 * k] + wv.y * hs[4 * k + 1] + wv.z * hs[4 * k + 2] + wv.w * hs[4 * k + 3];
  }
  if (cur == 0) acc = 0.0f;
  out[((size_t)t * BB + b) * NOO + o] = acc;
}

extern "C" void kernel_launch(void* const* d_in, const int* in_sizes, int n_in,
                              void* d_out, int out_size, void* d_ws, size_t ws_size,
                              hipStream_t stream) {
  (void)in_sizes; (void)n_in; (void)out_size; (void)ws_size;
  const int*   program       = (const int*)d_in[0];
  const int*   program_len   = (const int*)d_in[1];
  const int*   trace         = (const int*)d_in[2];
  const int*   choices       = (const int*)d_in[3];
  const float* enc_embed     = (const float*)d_in[4];
  const float* lstm_W_ih     = (const float*)d_in[5];
  const float* lstm_W_hh     = (const float*)d_in[6];
  const float* lstm_b_ih     = (const float*)d_in[7];
  const float* lstm_b_hh     = (const float*)d_in[8];
  const float* h0            = (const float*)d_in[9];
  const float* c0            = (const float*)d_in[10];
  const float* index_embed   = (const float*)d_in[11];
  const float* choice_tables = (const float*)d_in[12];
  const float* gru_W_ih      = (const float*)d_in[13];
  const float* gru_W_hh      = (const float*)d_in[14];
  const float* gru_b_ih      = (const float*)d_in[15];
  const float* gru_b_hh      = (const float*)d_in[16];
  const float* bn_gamma      = (const float*)d_in[17];
  const float* bn_beta       = (const float*)d_in[18];
  const float* inf_W         = (const float*)d_in[19];
  const float* inf_b         = (const float*)d_in[20];
  const float* h_dec0        = (const float*)d_in[21];
  float* out = (float*)d_out;

  // workspace carve-out (~6.7 MB total)
  char* base = (char*)d_ws;
  size_t off = 0;
  auto alloc = [&](size_t bytes) -> char* {
    char* p = base + off;
    off += (bytes + 255) & ~(size_t)255;
    return p;
  };
  float*    h_enc = (float*)alloc((size_t)BB * HH * 4);
  float*    c_enc = (float*)alloc((size_t)BB * HH * 4);
  float*    h_dec = (float*)alloc((size_t)BB * HH * 4);
  float*    hn32  = (float*)alloc((size_t)BB * HH * 4);
  _Float16* hn16  = (_Float16*)alloc((size_t)BB * HH * 2);
  _Float16* Aq    = (_Float16*)alloc((size_t)BB * KL * 2);
  _Float16* Iq    = (_Float16*)alloc((size_t)BB * KG * 2);
  _Float16* BpL   = (_Float16*)alloc((size_t)KL * NL * 2);
  _Float16* BpIH  = (_Float16*)alloc((size_t)KG * NG * 2);
  _Float16* BpHH  = (_Float16*)alloc((size_t)KH * NG * 2);

  // init recurrent states
  hipMemcpyAsync(h_enc, h0,     (size_t)BB * HH * 4, hipMemcpyDeviceToDevice, stream);
  hipMemcpyAsync(c_enc, c0,     (size_t)BB * HH * 4, hipMemcpyDeviceToDevice, stream);
  hipMemcpyAsync(h_dec, h_dec0, (size_t)BB * HH * 4, hipMemcpyDeviceToDevice, stream);

  // pre-pack weights to f16 WMMA B-fragment layout (stays L2-resident across steps)
  pack_b2_kernel<<<(KL * NL + 255) / 256, 256, 0, stream>>>(lstm_W_ih, EE, lstm_W_hh, HH, BpL, NL, KL);
  pack_b2_kernel<<<(KG * NG + 255) / 256, 256, 0, stream>>>(gru_W_ih, HH + 2 * EE, nullptr, 0, BpIH, NG, KG);
  pack_b2_kernel<<<(KH * NG + 255) / 256, 256, 0, stream>>>(gru_W_hh, HH, nullptr, 0, BpHH, NG, KH);

  // encoder: 256 length-masked LSTM steps
  for (int t = 0; t < PP; ++t) {
    pack_x_kernel<<<BB, 256, 0, stream>>>(enc_embed, program, h_enc, Aq, t);
    lstm_step_kernel<<<32, 256, 0, stream>>>(Aq, BpL, lstm_b_ih, lstm_b_hh, program_len, t, h_enc, c_enc);
  }

  // decoder: 127 BN + GRU + routed-head steps (prog_emb = final h_enc)
  for (int t = 0; t < TT - 1; ++t) {
    bn_kernel<<<HH / 8, 256, 0, stream>>>(h_dec, bn_gamma, bn_beta, hn32, hn16);
    pack_inp_kernel<<<BB, 256, 0, stream>>>(trace, choices, choice_tables, index_embed, h_enc, Iq, t);
    gru_step_kernel<<<32, 256, 0, stream>>>(Iq, BpIH, hn16, BpHH, hn32, gru_b_ih, gru_b_hh, h_dec);
    head_kernel<<<BB, NOO, 0, stream>>>(trace, h_dec, inf_W, inf_b, out, t);
  }
}
